// GAT_86071144612543
// MI455X (gfx1250) — matrix operand order repaired and checked by reading, not compiled, not run on hardware
//
#include <hip/hip_runtime.h>
#include <hip/hip_bf16.h>
#include <stdint.h>

typedef __attribute__((ext_vector_type(16))) _Float16 v16h;
typedef __attribute__((ext_vector_type(8)))  float    v8f;

#define NEG_SLOPE 0.2f

// ---------------------------------------------------------------------------
// float atomic max via CAS (portable across toolchains)
// ---------------------------------------------------------------------------
__device__ __forceinline__ void atomicMaxF(float* addr, float v) {
    unsigned int* ua = (unsigned int*)addr;
    unsigned int old = *ua;
    while (true) {
        float f = __uint_as_float(old);
        if (f >= v) break;
        unsigned int assumed = old;
        old = atomicCAS(ua, assumed, __float_as_uint(v));
        if (old == assumed) break;
    }
}

// ---------------------------------------------------------------------------
// f32 -> f16 cast (feature matrix for WMMA A operand)
// ---------------------------------------------------------------------------
__global__ void cast_f32_f16(const float* __restrict__ in, _Float16* __restrict__ out, int n) {
    int i = blockIdx.x * blockDim.x + threadIdx.x;
    if (i < n) out[i] = (_Float16)in[i];
}

// ---------------------------------------------------------------------------
// Pack W [K, ncols] f32 row-major into WMMA B-fragment order:
//   bp[((nt*ktiles + kt)*32 + lane)*8 + j] = {f16 W[k][col], f16 W[k+1][col]}
//   col = nt*16 + lane%16,  k = kt*32 + 2*j + 16*(lane/16)   (B 32x16 striping)
// Zero-pads columns >= ncols (layer 3: 7 real cols in a 16-wide tile).
// ---------------------------------------------------------------------------
__global__ void pack_b(const float* __restrict__ W, uint32_t* __restrict__ bp,
                       int Kdim, int ncols, int ntiles, int ktiles) {
    int idx = blockIdx.x * blockDim.x + threadIdx.x;
    int total = ntiles * ktiles * 32 * 8;
    if (idx >= total) return;
    int j    = idx & 7;
    int lane = (idx >> 3) & 31;
    int tk   = (idx >> 8);
    int kt   = tk % ktiles;
    int nt   = tk / ktiles;
    int col  = nt * 16 + (lane & 15);
    int h    = lane >> 4;
    int k    = kt * 32 + 2 * j + 16 * h;
    float w0 = 0.f, w1 = 0.f;
    if (col < ncols) {
        w0 = W[(size_t)k * ncols + col];
        w1 = W[(size_t)(k + 1) * ncols + col];
    }
    union { _Float16 h2[2]; uint32_t u; } p;
    p.h2[0] = (_Float16)w0;
    p.h2[1] = (_Float16)w1;
    bp[idx] = p.u;
}

// ---------------------------------------------------------------------------
// WMMA GEMM: one wave computes a full 16 x (NT*16) row-block of
//   C[nrows, ldc] = A[nrows, KT*32] (f16 row-major) x Bpack
// A fragment loaded once per K-step, reused across all NT column tiles.
// Fully unrolled: KT*NT v_wmma_f32_16x16x32_f16 per wave.
// ---------------------------------------------------------------------------
template <int KT, int NT>
__global__ void __launch_bounds__(128) gemm_wmma(
    const _Float16* __restrict__ A, const uint32_t* __restrict__ Bp,
    float* __restrict__ C, int nrows, int ldc, int ncols_store) {
    constexpr int K = KT * 32;
    int wave = threadIdx.x >> 5;
    int lane = threadIdx.x & 31;
    int mtiles = (nrows + 15) >> 4;
    int tm = blockIdx.x * 4 + wave;
    if (tm >= mtiles) return;                      // whole-wave uniform exit
    int l16 = lane & 15;
    int h   = lane >> 4;

    const _Float16* arp = A + (size_t)(tm * 16 + l16) * K;  // lane holds row lane%16

    v8f acc[NT] = {};
#pragma unroll
    for (int kt = 0; kt < KT; ++kt) {
        union { v16h v; uint32_t u[8]; } a;
#pragma unroll
        for (int j = 0; j < 8; ++j) {
            int k = kt * 32 + (j << 1) + (h << 4);
            a.u[j] = *(const uint32_t*)(arp + k);  // aligned pair of halves
        }
#pragma unroll
        for (int t = 0; t < NT; ++t) {
            union { v16h v; uint32_t u[8]; } b;
            const uint32_t* bb = Bp + (((size_t)t * KT + kt) * 32 + lane) * 8;
#pragma unroll
            for (int j = 0; j < 8; ++j) b.u[j] = bb[j];  // coalesced b128 loads
            acc[t] = __builtin_amdgcn_wmma_f32_16x16x32_f16(
                /*neg_a=*/false, a.v, /*neg_b=*/false, b.v,
                /*c_mod=*/(short)0, acc[t], /*reuse_a=*/false, /*reuse_b=*/false);
        }
    }

#pragma unroll
    for (int t = 0; t < NT; ++t) {
        int col = t * 16 + l16;
        if (col < ncols_store) {
#pragma unroll
            for (int r = 0; r < 8; ++r) {          // VGPR r: M = r + 8*(lane/16)
                int row = tm * 16 + h * 8 + r;
                if (row < nrows) C[(size_t)row * ldc + col] = acc[t][r];
            }
        }
    }
}

// ---------------------------------------------------------------------------
// Per-node: attention logit halves  as = <xp[n,h,:], a_src[h]>, ad likewise;
// also init segment max/sum and zero the message accumulator.
// Compile-time HEADS/HID -> fully unrolled, vectorizable loads.
// ---------------------------------------------------------------------------
template <int HEADS, int HID>
__global__ void node_prep(const float* __restrict__ xp,
                          const float* __restrict__ a_s, const float* __restrict__ a_d,
                          float* __restrict__ as_, float* __restrict__ ad_,
                          float* __restrict__ mbuf, float* __restrict__ sbuf,
                          float* __restrict__ acc, int n) {
    int i = blockIdx.x * blockDim.x + threadIdx.x;
    if (i >= n) return;
    constexpr int F = HEADS * HID;
    const float* row = xp + (size_t)i * F;
#pragma unroll
    for (int hd = 0; hd < HEADS; ++hd) {
        float ds = 0.f, dd = 0.f;
#pragma unroll
        for (int c = 0; c < HID; ++c) {
            float v = row[hd * HID + c];
            ds += v * a_s[hd * HID + c];
            dd += v * a_d[hd * HID + c];
        }
        as_[(size_t)i * HEADS + hd]  = ds;
        ad_[(size_t)i * HEADS + hd]  = dd;
        mbuf[(size_t)i * HEADS + hd] = -1e30f;
        sbuf[(size_t)i * HEADS + hd] = 0.f;
    }
#pragma unroll
    for (int j = 0; j < F; ++j) acc[(size_t)i * F + j] = 0.f;
}

// ---------------------------------------------------------------------------
// Edge pass 1: segment max of leaky_relu(as[src]+ad[dst]) over dst
// (edge i < E comes from edge_index; i >= E is the implicit self loop)
// ---------------------------------------------------------------------------
template <int HEADS>
__global__ void edge_max(const long long* __restrict__ ei, int E, int n,
                         const float* __restrict__ as_, const float* __restrict__ ad_,
                         float* __restrict__ mbuf) {
    int i = blockIdx.x * blockDim.x + threadIdx.x;
    if (i >= E + n) return;
    int src, dst;
    if (i < E) { src = (int)ei[i]; dst = (int)ei[(size_t)E + i]; }
    else       { src = dst = i - E; }
#pragma unroll
    for (int hd = 0; hd < HEADS; ++hd) {
        float e = as_[(size_t)src * HEADS + hd] + ad_[(size_t)dst * HEADS + hd];
        e = e > 0.f ? e : e * NEG_SLOPE;
        atomicMaxF(&mbuf[(size_t)dst * HEADS + hd], e);
    }
}

// ---------------------------------------------------------------------------
// Edge pass 2: segment sum of exp(e - max)
// ---------------------------------------------------------------------------
template <int HEADS>
__global__ void edge_expsum(const long long* __restrict__ ei, int E, int n,
                            const float* __restrict__ as_, const float* __restrict__ ad_,
                            const float* __restrict__ mbuf, float* __restrict__ sbuf) {
    int i = blockIdx.x * blockDim.x + threadIdx.x;
    if (i >= E + n) return;
    int src, dst;
    if (i < E) { src = (int)ei[i]; dst = (int)ei[(size_t)E + i]; }
    else       { src = dst = i - E; }
#pragma unroll
    for (int hd = 0; hd < HEADS; ++hd) {
        float e = as_[(size_t)src * HEADS + hd] + ad_[(size_t)dst * HEADS + hd];
        e = e > 0.f ? e : e * NEG_SLOPE;
        float p = __expf(e - mbuf[(size_t)dst * HEADS + hd]);
        atomicAdd(&sbuf[(size_t)dst * HEADS + hd], p);
    }
}

// ---------------------------------------------------------------------------
// Edge pass 3: out[dst] += alpha * xp[src]  (alpha = exp(e-m)/(s+eps))
// ---------------------------------------------------------------------------
template <int HEADS, int HID>
__global__ void edge_msg(const long long* __restrict__ ei, int E, int n,
                         const float* __restrict__ as_, const float* __restrict__ ad_,
                         const float* __restrict__ mbuf, const float* __restrict__ sbuf,
                         const float* __restrict__ xp, float* __restrict__ acc) {
    int i = blockIdx.x * blockDim.x + threadIdx.x;
    if (i >= E + n) return;
    int src, dst;
    if (i < E) { src = (int)ei[i]; dst = (int)ei[(size_t)E + i]; }
    else       { src = dst = i - E; }
    constexpr int F = HEADS * HID;
#pragma unroll
    for (int hd = 0; hd < HEADS; ++hd) {
        float e = as_[(size_t)src * HEADS + hd] + ad_[(size_t)dst * HEADS + hd];
        e = e > 0.f ? e : e * NEG_SLOPE;
        float alpha = __expf(e - mbuf[(size_t)dst * HEADS + hd]) /
                      (sbuf[(size_t)dst * HEADS + hd] + 1e-16f);
        const float* xrow = xp  + (size_t)src * F + hd * HID;
        float*       arow = acc + (size_t)dst * F + hd * HID;
#pragma unroll
        for (int c = 0; c < HID; ++c) atomicAdd(&arow[c], xrow[c] * alpha);
    }
}

// ---------------------------------------------------------------------------
// Finalize layers 1/2: h = relu(acc + bias) -> f16 (next layer's A operand)
// ---------------------------------------------------------------------------
__global__ void finalize_relu_f16(const float* __restrict__ acc, const float* __restrict__ bias,
                                  _Float16* __restrict__ out, int n, int F) {
    int i = blockIdx.x * blockDim.x + threadIdx.x;
    if (i >= n * F) return;
    int j = i % F;
    float v = acc[i] + bias[j];
    out[i] = (_Float16)(v > 0.f ? v : 0.f);
}

// ---------------------------------------------------------------------------
// Finalize layer 3: log_softmax(acc + bias) -> d_out (f32)
// ---------------------------------------------------------------------------
__global__ void finalize_logsoftmax(const float* __restrict__ acc, const float* __restrict__ bias,
                                    float* __restrict__ out, int n, int C) {
    int i = blockIdx.x * blockDim.x + threadIdx.x;
    if (i >= n) return;
    float l[8];
    float mx = -1e30f;
    for (int c = 0; c < C; ++c) {
        l[c] = acc[(size_t)i * C + c] + bias[c];
        mx = fmaxf(mx, l[c]);
    }
    float s = 0.f;
    for (int c = 0; c < C; ++c) s += __expf(l[c] - mx);
    float lse = __logf(s);
    for (int c = 0; c < C; ++c) out[(size_t)i * C + c] = l[c] - mx - lse;
}

// ---------------------------------------------------------------------------
extern "C" void kernel_launch(void* const* d_in, const int* in_sizes, int n_in,
                              void* d_out, int out_size, void* d_ws, size_t ws_size,
                              hipStream_t stream) {
    const float*     x   = (const float*)d_in[0];
    const long long* ei  = (const long long*)d_in[1];
    const float*     W1  = (const float*)d_in[2];
    const float*     a1s = (const float*)d_in[3];
    const float*     a1d = (const float*)d_in[4];
    const float*     b1  = (const float*)d_in[5];
    const float*     W2  = (const float*)d_in[6];
    const float*     a2s = (const float*)d_in[7];
    const float*     a2d = (const float*)d_in[8];
    const float*     b2  = (const float*)d_in[9];
    const float*     W3  = (const float*)d_in[10];
    const float*     a3s = (const float*)d_in[11];
    const float*     a3d = (const float*)d_in[12];
    const float*     b3  = (const float*)d_in[13];

    const int F_IN = 128, F = 64, C3 = 7;
    int N = in_sizes[0] / F_IN;
    int E = in_sizes[1] / 2;
    int Et = E + N;

    // ---- workspace carving (256B-aligned) ----
    char* p = (char*)d_ws;
    auto carve = [&](size_t bytes) -> void* {
        void* r = (void*)p;
        p += (bytes + 255) & ~(size_t)255;
        return r;
    };
    _Float16* xf16 = (_Float16*)carve((size_t)N * F_IN * sizeof(_Float16));
    _Float16* hf16 = (_Float16*)carve((size_t)N * F * sizeof(_Float16));
    float*    xp   = (float*)carve((size_t)N * F * sizeof(float));
    float*    acc  = (float*)carve((size_t)N * F * sizeof(float));
    float*    as_  = (float*)carve((size_t)N * 8 * sizeof(float));
    float*    ad_  = (float*)carve((size_t)N * 8 * sizeof(float));
    float*    mb   = (float*)carve((size_t)N * 8 * sizeof(float));
    float*    sb   = (float*)carve((size_t)N * 8 * sizeof(float));
    uint32_t* b1p  = (uint32_t*)carve((size_t)4 * 4 * 256 * 4);
    uint32_t* b2p  = (uint32_t*)carve((size_t)4 * 2 * 256 * 4);
    uint32_t* b3p  = (uint32_t*)carve((size_t)1 * 2 * 256 * 4);

    const int T = 256;
    int mtiles = (N + 15) / 16;
    int gblk   = (mtiles + 3) / 4;      // 4 waves per block, 1 row-block per wave

    // feature cast + weight packing (cheap, once per launch)
    cast_f32_f16<<<(N * F_IN + T - 1) / T, T, 0, stream>>>(x, xf16, N * F_IN);
    pack_b<<<(4 * 4 * 256 + T - 1) / T, T, 0, stream>>>(W1, b1p, 128, 64, 4, 4);
    pack_b<<<(4 * 2 * 256 + T - 1) / T, T, 0, stream>>>(W2, b2p, 64, 64, 4, 2);
    pack_b<<<(1 * 2 * 256 + T - 1) / T, T, 0, stream>>>(W3, b3p, 64, 7, 1, 2);

    // ---------------- Layer 1: 128 -> 8x8, concat ----------------
    gemm_wmma<4, 4><<<gblk, 128, 0, stream>>>(xf16, b1p, xp, N, 64, 64);
    node_prep<8, 8><<<(N + T - 1) / T, T, 0, stream>>>(xp, a1s, a1d, as_, ad_, mb, sb, acc, N);
    edge_max<8>   <<<(Et + T - 1) / T, T, 0, stream>>>(ei, E, N, as_, ad_, mb);
    edge_expsum<8><<<(Et + T - 1) / T, T, 0, stream>>>(ei, E, N, as_, ad_, mb, sb);
    edge_msg<8, 8><<<(Et + T - 1) / T, T, 0, stream>>>(ei, E, N, as_, ad_, mb, sb, xp, acc);
    finalize_relu_f16<<<(N * F + T - 1) / T, T, 0, stream>>>(acc, b1, hf16, N, F);

    // ---------------- Layer 2: 64 -> 8x8, concat ----------------
    gemm_wmma<2, 4><<<gblk, 128, 0, stream>>>(hf16, b2p, xp, N, 64, 64);
    node_prep<8, 8><<<(N + T - 1) / T, T, 0, stream>>>(xp, a2s, a2d, as_, ad_, mb, sb, acc, N);
    edge_max<8>   <<<(Et + T - 1) / T, T, 0, stream>>>(ei, E, N, as_, ad_, mb);
    edge_expsum<8><<<(Et + T - 1) / T, T, 0, stream>>>(ei, E, N, as_, ad_, mb, sb);
    edge_msg<8, 8><<<(Et + T - 1) / T, T, 0, stream>>>(ei, E, N, as_, ad_, mb, sb, xp, acc);
    finalize_relu_f16<<<(N * F + T - 1) / T, T, 0, stream>>>(acc, b2, hf16, N, F);

    // ---------------- Layer 3: 64 -> 1x7, mean (=identity for 1 head) ------
    gemm_wmma<2, 1><<<gblk, 128, 0, stream>>>(hf16, b3p, xp, N, C3, C3);
    node_prep<1, 7><<<(N + T - 1) / T, T, 0, stream>>>(xp, a3s, a3d, as_, ad_, mb, sb, acc, N);
    edge_max<1>   <<<(Et + T - 1) / T, T, 0, stream>>>(ei, E, N, as_, ad_, mb);
    edge_expsum<1><<<(Et + T - 1) / T, T, 0, stream>>>(ei, E, N, as_, ad_, mb, sb);
    edge_msg<1, 7><<<(Et + T - 1) / T, T, 0, stream>>>(ei, E, N, as_, ad_, mb, sb, xp, acc);
    finalize_logsoftmax<<<(N + T - 1) / T, T, 0, stream>>>(acc, b3, (float*)d_out, N, C3);
}
